// MultiHeadAttention_34076270526864
// MI455X (gfx1250) — matrix operand order repaired and checked by reading, not compiled, not run on hardware
//
#include <hip/hip_runtime.h>
#include <cmath>

typedef __attribute__((ext_vector_type(16))) __bf16 v16bf;
typedef __attribute__((ext_vector_type(8)))  __bf16 v8bf;
typedef __attribute__((ext_vector_type(8)))  float  v8f;

#define D_MODEL 512
#define SEQ     2048
#define BATCH   4
#define HEADS   8
#define DHEAD   64

#ifndef __has_builtin
#define __has_builtin(x) 0
#endif

#if defined(__HIP_DEVICE_COMPILE__)
#if __has_builtin(__builtin_amdgcn_tensor_load_to_lds) && \
    __has_builtin(__builtin_amdgcn_s_wait_tensorcnt)
#define HAVE_TDM 1
#if __has_include(<hip/amd_detail/amd_gfx1250_TDM.h>)
#define TDM_6ARG 1
#endif
#endif
#endif

static __device__ __forceinline__ v16bf cat8(v8bf lo, v8bf hi) {
  return __builtin_shufflevector(lo, hi, 0,1,2,3,4,5,6,7,8,9,10,11,12,13,14,15);
}
static __device__ __forceinline__ v8f zero8() {
  v8f z = {0.f,0.f,0.f,0.f,0.f,0.f,0.f,0.f};
  return z;
}
static __device__ __forceinline__ v8f wmma_bf16(v16bf a, v16bf b, v8f c) {
  return __builtin_amdgcn_wmma_f32_16x16x32_bf16(false, a, false, b, (short)0, c, false, false);
}

#if defined(HAVE_TDM)
typedef __attribute__((ext_vector_type(4))) unsigned int u32x4;
typedef __attribute__((ext_vector_type(4))) int          i32x4;
typedef __attribute__((ext_vector_type(8))) int          i32x8;

// 2D tensor_load_to_lds: D# per CDNA5 ISA.
static __device__ __forceinline__ void tdm_load_2d(
    const void* gaddr, unsigned lds_addr,
    unsigned tile0, unsigned tile1, unsigned stride0,
    unsigned tdim0, unsigned tdim1) {
  unsigned long long ga = (unsigned long long)(size_t)gaddr;
  u32x4 g0 = { 1u,                                   // count=1, user mode
               lds_addr,                             // lds_addr [63:32]
               (unsigned)ga,                         // global_addr [95:64]
               (unsigned)((ga >> 32) & 0x01ffffffu)  // global_addr [120:96]
               | 0x80000000u };                      // type=2 @ [127:126]
  i32x8 g1 = { (int)(1u << 16),                                       // data_size=2B
               (int)((tdim0 & 0xffffu) << 16),                        // tensor_dim0 lo
               (int)(((tdim0 >> 16) & 0xffffu) | ((tdim1 & 0xffffu) << 16)),
               (int)(((tdim1 >> 16) & 0xffffu) | (tile0 << 16)),      // tile_dim0
               (int)(tile1 & 0xffffu),                                // tile_dim1, tile_dim2=0
               (int)stride0,                                          // dim0 stride lo32
               0, 0 };
  i32x4 z4 = {0, 0, 0, 0};
#if defined(TDM_6ARG)
  i32x8 z8 = {0, 0, 0, 0, 0, 0, 0, 0};
  __builtin_amdgcn_tensor_load_to_lds(g0, g1, z4, z4, z8, 0);
#else
  __builtin_amdgcn_tensor_load_to_lds(g0, g1, z4, z4, 0);
#endif
}
#endif  // HAVE_TDM

// ---------------------------------------------------------------------------
// Kernel 1: positional encoding + add + convert to bf16 (both streams at once)
// ---------------------------------------------------------------------------
__global__ void __launch_bounds__(256) pe_add_cvt_kernel(
    const float* __restrict__ xp, const float* __restrict__ xm,
    __bf16* __restrict__ op, __bf16* __restrict__ om) {
  size_t idx = (size_t)blockIdx.x * 256 + threadIdx.x;      // covers B*S*D exactly
  int d = (int)(idx & (D_MODEL - 1));
  int s = (int)((idx >> 9) & (SEQ - 1));
  int i = d >> 1;
  float ang = (float)s * __expf((float)i * 0.0359778920f);  // ln(10000)/256
  float pe  = (d & 1) ? __cosf(ang) : __sinf(ang);
  op[idx] = (__bf16)(xp[idx] + pe);
  om[idx] = (__bf16)(xm[idx] + pe);
}

// ---------------------------------------------------------------------------
// Kernel 2: f32 -> bf16 weight convert (layout preserved: [N=out, K=in])
// ---------------------------------------------------------------------------
__global__ void __launch_bounds__(256) wcvt_kernel(const float* __restrict__ w,
                                                   __bf16* __restrict__ wb) {
  size_t idx = (size_t)blockIdx.x * 256 + threadIdx.x;
  wb[idx] = (__bf16)w[idx];
}

// ---------------------------------------------------------------------------
// Kernel 3: WMMA GEMM   C[m,n] = sum_k A[m,k] * W[n,k]
//   A: [8192,512] bf16 row-major, W: [512,512] bf16 row-major.
//   Block = 256 threads (8 waves), tile 128(M) x 64(N); wave: 16 x 64.
//   TDM path: A (128x64, 16 KB) and W (64x64, 8 KB) tiles double-buffered in
//   LDS; inner loop entirely LDS-fed (2 K-substeps of 32 per stage).
//   MODE 0: bf16 [M,512] | MODE 1: f32 [M,512] | MODE 2: bf16 VT [B,H,64,S]
// ---------------------------------------------------------------------------
template <int MODE>
__global__ void __launch_bounds__(256) gemm_bf16_kernel(
    const __bf16* __restrict__ A, const __bf16* __restrict__ W,
    void* __restrict__ out) {
  const int tid  = threadIdx.x;
  const int wave = tid >> 5;
  const int lane = tid & 31;
  const int half = lane >> 4;
  const int l15  = lane & 15;
  const int mblk = blockIdx.x * 128;
  const int nblk = blockIdx.y * 64;

  v8f acc[4];
#pragma unroll
  for (int t = 0; t < 4; ++t) acc[t] = zero8();

#if defined(HAVE_TDM)
  __shared__ __bf16 As[2][128 * 64];                        // 2 x 16 KB
  __shared__ __bf16 Wt[2][64 * 64];                         // 2 x 8 KB
  const int NSTG = D_MODEL / 64;                            // 8 stages
  if (tid == 0) {
    tdm_load_2d(A + (size_t)mblk * D_MODEL, (unsigned)(size_t)&As[0][0],
                64, 128, D_MODEL, D_MODEL, BATCH * SEQ);
    tdm_load_2d(W + (size_t)nblk * D_MODEL, (unsigned)(size_t)&Wt[0][0],
                64, 64, D_MODEL, D_MODEL, D_MODEL);
  }
  for (int stg = 0; stg < NSTG; ++stg) {
    const int cur = stg & 1;
    if (tid == 0) {
      if (stg + 1 < NSTG) {
        tdm_load_2d(A + (size_t)mblk * D_MODEL + (stg + 1) * 64,
                    (unsigned)(size_t)&As[cur ^ 1][0],
                    64, 128, D_MODEL, D_MODEL, BATCH * SEQ);
        tdm_load_2d(W + (size_t)nblk * D_MODEL + (stg + 1) * 64,
                    (unsigned)(size_t)&Wt[cur ^ 1][0],
                    64, 64, D_MODEL, D_MODEL, D_MODEL);
        __builtin_amdgcn_s_wait_tensorcnt(2);               // current A+W landed
      } else {
        __builtin_amdgcn_s_wait_tensorcnt(0);
      }
    }
    __syncthreads();
#pragma unroll
    for (int sub = 0; sub < 2; ++sub) {
      const __bf16* arow = &As[cur][(wave * 16 + l15) * 64 + sub * 32];
      v16bf af = *(const v16bf*)(arow + 16 * half);
#pragma unroll
      for (int t = 0; t < 4; ++t) {
        const __bf16* wp = &Wt[cur][(t * 16 + l15) * 64 + sub * 32 + 16 * half];
        acc[t] = wmma_bf16(af, *(const v16bf*)wp, acc[t]);
      }
    }
    __syncthreads();
  }
#else
  __shared__ __bf16 As[128 * 32];                           // 8 KB A tile
  for (int k0 = 0; k0 < D_MODEL; k0 += 32) {
    {
      int row = tid >> 1;
      int off = (tid & 1) * 16;                             // halves
      const __bf16* src = A + (size_t)(mblk + row) * D_MODEL + k0 + off;
      __bf16* dst = &As[row * 32 + off];
      *(uint4*)(dst)     = *(const uint4*)(src);
      *(uint4*)(dst + 8) = *(const uint4*)(src + 8);
    }
    __syncthreads();
    const __bf16* arow = &As[(wave * 16 + l15) * 32];
    v16bf af = *(const v16bf*)(arow + 16 * half);
#pragma unroll
    for (int t = 0; t < 4; ++t) {
      const __bf16* wp = W + (size_t)(nblk + t * 16 + l15) * D_MODEL + k0 + 16 * half;
      acc[t] = wmma_bf16(af, *(const v16bf*)wp, acc[t]);
    }
    __syncthreads();
  }
#endif

  // epilogue: C layout -> row m = v + 8*half, col = l15 within each 16x16 tile
#pragma unroll
  for (int t = 0; t < 4; ++t) {
#pragma unroll
    for (int v = 0; v < 8; ++v) {
      int row = mblk + wave * 16 + v + 8 * half;
      int col = nblk + t * 16 + l15;
      float val = acc[t][v];
      if constexpr (MODE == 0) {
        ((__bf16*)out)[(size_t)row * D_MODEL + col] = (__bf16)val;
      } else if constexpr (MODE == 1) {
        ((float*)out)[(size_t)row * D_MODEL + col] = val;
      } else {                                              // VT scatter
        int b = row >> 11, s = row & (SEQ - 1);
        int h = col >> 6,  dh = col & (DHEAD - 1);
        ((__bf16*)out)[(((size_t)b * HEADS + h) * DHEAD + dh) * SEQ + s] = (__bf16)val;
      }
    }
  }
}

// ---------------------------------------------------------------------------
// Kernel 4: flash attention per (b, h).
//   Q,K: bf16 [B,S,512] (head h = cols h*64..);  VT: bf16 [B,H,64,S].
//   Block = 256 threads (8 waves); wave owns 16 q rows.
//   TDM path: K (64kv x 64dh) and V (64dh x 64kv) tiles double-buffered in
//   LDS, shared by all 8 waves; 2 kv-substeps of 32 per stage.
// ---------------------------------------------------------------------------
__global__ void __launch_bounds__(256) attn_kernel(
    const __bf16* __restrict__ Q, const __bf16* __restrict__ K,
    const __bf16* __restrict__ VT, __bf16* __restrict__ Xa) {
  __shared__ __bf16 Pld[8 * 16 * 32];                       // 1 KB per wave (P reshape)
#if defined(HAVE_TDM)
  __shared__ __bf16 Kt[2][64 * 64];                         // kv x dh, 2 x 8 KB
  __shared__ __bf16 Vtl[2][64 * 64];                        // dh x kv, 2 x 8 KB
#endif

  const int wave = threadIdx.x >> 5;
  const int lane = threadIdx.x & 31;
  const int half = lane >> 4;
  const int l15  = lane & 15;
  const int b = blockIdx.z, h = blockIdx.y;
  const int q0 = blockIdx.x * 128 + wave * 16;
  const float SCALE = 0.125f;                               // 1/sqrt(64)
  const int NSTG = SEQ / 64;                                // 32 kv stages

  const __bf16* Qh = Q  + (size_t)b * SEQ * D_MODEL + h * DHEAD;
  const __bf16* Kh = K  + (size_t)b * SEQ * D_MODEL + h * DHEAD;
  const __bf16* Vt = VT + ((size_t)b * HEADS + h) * DHEAD * SEQ;

  // Q fragments for both K-steps (dh 0..31 / 32..63), resident all loop
  const __bf16* qrow = Qh + (size_t)(q0 + l15) * D_MODEL;
  v16bf qf0 = *(const v16bf*)(qrow + 16 * half);
  v16bf qf1 = *(const v16bf*)(qrow + 32 + 16 * half);

  v8f o[4];
#pragma unroll
  for (int t = 0; t < 4; ++t) o[t] = zero8();
  float m_run[8], l_run[8];
#pragma unroll
  for (int v = 0; v < 8; ++v) { m_run[v] = -3.0e38f; l_run[v] = 0.f; }

  __bf16* pw = &Pld[wave * 512];

#if defined(HAVE_TDM)
  if (threadIdx.x == 0) {
    tdm_load_2d(Kh, (unsigned)(size_t)&Kt[0][0],  64, 64, D_MODEL, D_MODEL, BATCH * SEQ);
    tdm_load_2d(Vt, (unsigned)(size_t)&Vtl[0][0], 64, 64, SEQ, SEQ, BATCH * HEADS * DHEAD);
  }
#endif

  for (int stg = 0; stg < NSTG; ++stg) {
#if defined(HAVE_TDM)
    const int cur = stg & 1;
    if (threadIdx.x == 0) {
      if (stg + 1 < NSTG) {
        tdm_load_2d(Kh + (size_t)(stg + 1) * 64 * D_MODEL,
                    (unsigned)(size_t)&Kt[cur ^ 1][0],
                    64, 64, D_MODEL, D_MODEL, BATCH * SEQ);
        tdm_load_2d(Vt + (stg + 1) * 64, (unsigned)(size_t)&Vtl[cur ^ 1][0],
                    64, 64, SEQ, SEQ, BATCH * HEADS * DHEAD);
        __builtin_amdgcn_s_wait_tensorcnt(2);               // current K+V landed
      } else {
        __builtin_amdgcn_s_wait_tensorcnt(0);
      }
    }
    __syncthreads();
#endif

#pragma unroll
    for (int sub = 0; sub < 2; ++sub) {
      v16bf kb00, kb01, kb10, kb11, vfr0, vfr1, vfr2, vfr3;
#if defined(HAVE_TDM)
      const __bf16* kb = &Kt[cur][sub * 32 * 64];
      kb00 = *(const v16bf*)(kb + l15 * 64 + 16 * half);
      kb01 = *(const v16bf*)(kb + l15 * 64 + 32 + 16 * half);
      kb10 = *(const v16bf*)(kb + (16 + l15) * 64 + 16 * half);
      kb11 = *(const v16bf*)(kb + (16 + l15) * 64 + 32 + 16 * half);
      const __bf16* vb = &Vtl[cur][sub * 32];
      vfr0 = *(const v16bf*)(vb + (l15) * 64 + 16 * half);
      vfr1 = *(const v16bf*)(vb + (16 + l15) * 64 + 16 * half);
      vfr2 = *(const v16bf*)(vb + (32 + l15) * 64 + 16 * half);
      vfr3 = *(const v16bf*)(vb + (48 + l15) * 64 + 16 * half);
#else
      const int kv0 = stg * 64 + sub * 32;
      const __bf16* kra = Kh + (size_t)(kv0 + l15) * D_MODEL + 16 * half;
      const __bf16* krb = Kh + (size_t)(kv0 + 16 + l15) * D_MODEL + 16 * half;
      kb00 = *(const v16bf*)(kra);
      kb01 = *(const v16bf*)(kra + 32);
      kb10 = *(const v16bf*)(krb);
      kb11 = *(const v16bf*)(krb + 32);
      vfr0 = *(const v16bf*)(Vt + (size_t)(l15) * SEQ + kv0 + 16 * half);
      vfr1 = *(const v16bf*)(Vt + (size_t)(16 + l15) * SEQ + kv0 + 16 * half);
      vfr2 = *(const v16bf*)(Vt + (size_t)(32 + l15) * SEQ + kv0 + 16 * half);
      vfr3 = *(const v16bf*)(Vt + (size_t)(48 + l15) * SEQ + kv0 + 16 * half);
#endif

      // scores: two 16x16 kv tiles, K-dim 64 => 2 WMMAs each
      v8f c0 = zero8(), c1 = zero8();
      c0 = wmma_bf16(qf0, kb00, c0);
      c0 = wmma_bf16(qf1, kb01, c0);
      c1 = wmma_bf16(qf0, kb10, c1);
      c1 = wmma_bf16(qf1, kb11, c1);

      // online softmax over the 32 kv columns
      float s0v[8], s1v[8], rm[8], alpha[8], p0v[8], p1v[8], rs[8];
#pragma unroll
      for (int v = 0; v < 8; ++v) {
        s0v[v] = c0[v] * SCALE;
        s1v[v] = c1[v] * SCALE;
        rm[v]  = fmaxf(s0v[v], s1v[v]);
      }
#pragma unroll
      for (int off = 8; off >= 1; off >>= 1)
#pragma unroll
        for (int v = 0; v < 8; ++v)
          rm[v] = fmaxf(rm[v], __shfl_xor(rm[v], off, 32));
#pragma unroll
      for (int v = 0; v < 8; ++v) {
        float mn = fmaxf(m_run[v], rm[v]);
        alpha[v] = __expf(m_run[v] - mn);
        m_run[v] = mn;
        p0v[v] = __expf(s0v[v] - mn);
        p1v[v] = __expf(s1v[v] - mn);
        rs[v]  = p0v[v] + p1v[v];
      }
#pragma unroll
      for (int off = 8; off >= 1; off >>= 1)
#pragma unroll
        for (int v = 0; v < 8; ++v)
          rs[v] += __shfl_xor(rs[v], off, 32);
#pragma unroll
      for (int v = 0; v < 8; ++v)
        l_run[v] = l_run[v] * alpha[v] + rs[v];
#pragma unroll
      for (int t = 0; t < 4; ++t)
#pragma unroll
        for (int v = 0; v < 8; ++v)
          o[t][v] *= alpha[v];

      // reshape P from C-layout to A-layout through per-wave LDS region
#pragma unroll
      for (int v = 0; v < 8; ++v) {
        pw[(v + 8 * half) * 32 + l15]      = (__bf16)p0v[v];
        pw[(v + 8 * half) * 32 + 16 + l15] = (__bf16)p1v[v];
      }
      asm volatile("s_wait_dscnt 0" ::: "memory");          // intra-wave LDS RAW

      const __bf16* pr = &Pld[wave * 512 + l15 * 32];
      v16bf pf = *(const v16bf*)(pr + 16 * half);

      // O += P(16x32) @ V(32x64)
      o[0] = wmma_bf16(pf, vfr0, o[0]);
      o[1] = wmma_bf16(pf, vfr1, o[1]);
      o[2] = wmma_bf16(pf, vfr2, o[2]);
      o[3] = wmma_bf16(pf, vfr3, o[3]);
    }

#if defined(HAVE_TDM)
    __syncthreads();                                        // before buffer reuse
#endif
  }

  // normalize and store bf16 into Xa[b, q, h*64 + dh]
#pragma unroll
  for (int v = 0; v < 8; ++v) {
    float inv = 1.0f / l_run[v];
    int row = q0 + v + 8 * half;
#pragma unroll
    for (int t = 0; t < 4; ++t) {
      int col = h * DHEAD + t * 16 + l15;
      Xa[((size_t)b * SEQ + row) * D_MODEL + col] = (__bf16)(o[t][v] * inv);
    }
  }
}

// ---------------------------------------------------------------------------
extern "C" void kernel_launch(void* const* d_in, const int* in_sizes, int n_in,
                              void* d_out, int out_size, void* d_ws, size_t ws_size,
                              hipStream_t stream) {
  (void)in_sizes; (void)n_in; (void)out_size; (void)ws_size;
  const float* memory_p = (const float*)d_in[0];
  const float* memory   = (const float*)d_in[1];
  const float* Wq = (const float*)d_in[2];
  const float* Wk = (const float*)d_in[3];
  const float* Wv = (const float*)d_in[4];
  const float* Wo = (const float*)d_in[5];

  const size_t ACT  = (size_t)BATCH * SEQ * D_MODEL * sizeof(__bf16);  // 8 MB
  const size_t WSZ  = (size_t)D_MODEL * D_MODEL * sizeof(__bf16);      // 0.5 MB
  char* ws = (char*)d_ws;
  __bf16* Xp  = (__bf16*)(ws);                 ws += ACT;
  __bf16* Xm  = (__bf16*)(ws);                 ws += ACT;
  __bf16* Wqb = (__bf16*)(ws);                 ws += WSZ;
  __bf16* Wkb = (__bf16*)(ws);                 ws += WSZ;
  __bf16* Wvb = (__bf16*)(ws);                 ws += WSZ;
  __bf16* Wob = (__bf16*)(ws);                 ws += WSZ;
  __bf16* Qb  = (__bf16*)(ws);                 ws += ACT;
  __bf16* Kb  = (__bf16*)(ws);                 ws += ACT;
  __bf16* VT  = (__bf16*)(ws);                 ws += ACT;
  __bf16* Xa  = (__bf16*)(ws);                 ws += ACT;

  // 1) positional encoding + bf16 convert
  pe_add_cvt_kernel<<<(BATCH * SEQ * D_MODEL) / 256, 256, 0, stream>>>(
      memory_p, memory, Xp, Xm);

  // 2) weight converts
  wcvt_kernel<<<(D_MODEL * D_MODEL) / 256, 256, 0, stream>>>(Wq, Wqb);
  wcvt_kernel<<<(D_MODEL * D_MODEL) / 256, 256, 0, stream>>>(Wk, Wkb);
  wcvt_kernel<<<(D_MODEL * D_MODEL) / 256, 256, 0, stream>>>(Wv, Wvb);
  wcvt_kernel<<<(D_MODEL * D_MODEL) / 256, 256, 0, stream>>>(Wo, Wob);

  // 3) projections (M=8192 rows, 64 M-blocks x 8 N-blocks)
  dim3 ggrid(BATCH * SEQ / 128, D_MODEL / 64);
  gemm_bf16_kernel<0><<<ggrid, 256, 0, stream>>>(Xp, Wqb, Qb);
  gemm_bf16_kernel<0><<<ggrid, 256, 0, stream>>>(Xm, Wkb, Kb);
  gemm_bf16_kernel<2><<<ggrid, 256, 0, stream>>>(Xm, Wvb, VT);

  // 4) attention
  attn_kernel<<<dim3(SEQ / 128, HEADS, BATCH), 256, 0, stream>>>(Qb, Kb, VT, Xa);

  // 5) output projection, f32 result
  gemm_bf16_kernel<1><<<ggrid, 256, 0, stream>>>(Xa, Wob, (float*)d_out);
}
